// SimpleHGT_31121333026890
// MI455X (gfx1250) — matrix operand order repaired and checked by reading, not compiled, not run on hardware
//
#include <hip/hip_runtime.h>
#include <hip/hip_bf16.h>

typedef __attribute__((ext_vector_type(16))) __bf16 v16bf;
typedef __attribute__((ext_vector_type(8)))  __bf16 v8bf;
typedef __attribute__((ext_vector_type(8)))  float  v8f;

#define HEADS 2
#define DH    32
#define HID   64
#define FIN   32

// ---------------------------------------------------------------------------
// elementwise f32 -> bf16 convert
// ---------------------------------------------------------------------------
__global__ void k_f32_to_bf16(const float* __restrict__ in, __bf16* __restrict__ out, int n) {
  int i = blockIdx.x * blockDim.x + threadIdx.x;
  if (i < n) out[i] = (__bf16)in[i];
}

// ---------------------------------------------------------------------------
// WMMA GEMM: C[M,N] = A[M,K](bf16, leading dim LDA) * B[K,N](f32, ldb==N) + bias
//  - compile-time K/N/LDA/LDC -> all addressing is immediate offsets
//  - B staged once per block into LDS, transposed + converted to bf16; the
//    per-lane B fragment is then two 16B ds_load_b128 (K-contiguous)
//  - one wave computes a 16 x N row-block: A fragment reused across N/16 WMMAs
//  - requires M%16==0 (all call sites), K%32==0, N%16==0
// ---------------------------------------------------------------------------
template <int K, int N, int LDA, int LDC, bool BIAS, bool WF32, bool WBF16>
__global__ void k_gemm_wmma(const __bf16* __restrict__ A,
                            const float* __restrict__ B,
                            const float* __restrict__ bias,
                            float* __restrict__ Cf, __bf16* __restrict__ Cbf,
                            int M) {
  constexpr int NT = N / 16;                    // 16x16 tiles per row-block
  __shared__ __align__(16) __bf16 ldsBt[K * N]; // B transposed: [col][k], bf16

  // ---- stage B (uniform across block, before any divergence) -------------
  for (int i = threadIdx.x; i < K * N; i += blockDim.x) {
    int kk = i / N, col = i % N;                // coalesced global read
    ldsBt[col * K + kk] = (__bf16)B[i];
  }
  __syncthreads();

  const int lane = threadIdx.x & 31;
  const int wave = threadIdx.x >> 5;
  const int mt   = blockIdx.x * (blockDim.x >> 5) + wave;
  if (mt >= (M >> 4)) return;

  const int l15 = lane & 15;
  const int kg  = lane >> 4;                    // K-group select (0/1)
  const int row = (mt << 4) + l15;

  v8f acc[NT];
#pragma unroll
  for (int t = 0; t < NT; ++t) acc[t] = (v8f){};

#pragma unroll
  for (int kb = 0; kb < K; kb += 32) {
    // A fragment: K = kb + kg*8 + {0..7}  and  kb + 16 + kg*8 + {0..7}
    const __bf16* ap = A + (size_t)row * LDA + kb + (kg << 3);
    v8bf alo = *(const v8bf*)ap;
    v8bf ahi = *(const v8bf*)(ap + 16);
    v16bf a;
#pragma unroll
    for (int e = 0; e < 8; ++e) { a[e] = alo[e]; a[e + 8] = ahi[e]; }

#pragma unroll
    for (int t = 0; t < NT; ++t) {
      // B fragment: col = t*16 + l15, K = kb + kg*8 + {0..7} and kb + 16 + kg*8 + {0..7}
      const __bf16* bp = &ldsBt[(t * 16 + l15) * K + kb + (kg << 3)];
      v8bf blo = *(const v8bf*)bp;
      v8bf bhi = *(const v8bf*)(bp + 16);
      v16bf b;
#pragma unroll
      for (int e = 0; e < 8; ++e) { b[e] = blo[e]; b[e + 8] = bhi[e]; }
      acc[t] = __builtin_amdgcn_wmma_f32_16x16x32_bf16(false, a, false, b,
                                                       (short)0, acc[t], false, false);
    }
  }

  // ---- epilogue: C row = vgpr + 8*kg + 16*mt, col = t*16 + l15 -----------
  const int rbase = (mt << 4) + (kg << 3);
#pragma unroll
  for (int t = 0; t < NT; ++t) {
    const int col = (t << 4) + l15;
    float bv = 0.0f;
    if constexpr (BIAS) bv = bias[col];
    if constexpr (WF32) {
      float* p = Cf + (size_t)rbase * LDC + col;
#pragma unroll
      for (int v = 0; v < 8; ++v) { *p = acc[t][v] + bv; p += LDC; }
    }
    if constexpr (WBF16) {
      __bf16* p = Cbf + (size_t)rbase * LDC + col;
#pragma unroll
      for (int v = 0; v < 8; ++v) { *p = (__bf16)(acc[t][v] + bv); p += LDC; }
    }
  }
}

// ---------------------------------------------------------------------------
// Segment softmax over incoming edges (dst = flow nodes), 3 relations.
// Ordered-uint encoding so atomicMax works for signed floats.
// ---------------------------------------------------------------------------
__device__ __forceinline__ unsigned ord_enc(float f) {
  unsigned u = __float_as_uint(f);
  return (u & 0x80000000u) ? ~u : (u | 0x80000000u);
}
__device__ __forceinline__ float ord_dec(unsigned u) {
  return __uint_as_float((u & 0x80000000u) ? (u ^ 0x80000000u) : ~u);
}

__global__ void k_edge_logit(const int* __restrict__ src, const int* __restrict__ dst,
                             const float* __restrict__ qf, const float* __restrict__ kr,
                             const float* __restrict__ prel,
                             float* __restrict__ logit, unsigned* __restrict__ maxb, int nE) {
  int i = blockIdx.x * blockDim.x + threadIdx.x;
  if (i >= nE * 2) return;
  int e = i >> 1, h = i & 1;
  int s = src[e], d = dst[e];
  const float4* qp = (const float4*)(qf + (size_t)d * HID + h * DH);
  const float4* kp = (const float4*)(kr + (size_t)s * HID + h * DH);
  float acc = 0.0f;
#pragma unroll
  for (int j = 0; j < 8; ++j) {
    float4 a = qp[j], b = kp[j];
    acc += a.x * b.x + a.y * b.y + a.z * b.z + a.w * b.w;
  }
  float lg = acc * prel[h] * 0.17677669529663687f;   // 1/sqrt(32)
  logit[(size_t)e * 2 + h] = lg;
  atomicMax(maxb + (size_t)d * 2 + h, ord_enc(lg));
}

__global__ void k_edge_exp(const int* __restrict__ dst, float* __restrict__ logit,
                           const unsigned* __restrict__ maxb, float* __restrict__ sumb,
                           int nE) {
  int i = blockIdx.x * blockDim.x + threadIdx.x;
  if (i >= nE * 2) return;
  int e = i >> 1, h = i & 1;
  int d = dst[e];
  float m  = ord_dec(maxb[(size_t)d * 2 + h]);
  float ex = __expf(logit[(size_t)e * 2 + h] - m);
  logit[(size_t)e * 2 + h] = ex;
  atomicAdd(sumb + (size_t)d * 2 + h, ex);
}

__global__ void k_edge_agg(const int* __restrict__ src, const int* __restrict__ dst,
                           const float* __restrict__ logit, const float* __restrict__ sumb,
                           const float* __restrict__ vm, float* __restrict__ agg, int nE) {
  long long i = (long long)blockIdx.x * blockDim.x + threadIdx.x;
  if (i >= (long long)nE * HID) return;
  int e = (int)(i >> 6);
  int c = (int)(i & 63);
  int h = c >> 5;
  int s = src[e], d = dst[e];
  float alpha = logit[(size_t)e * 2 + h] / (sumb[(size_t)d * 2 + h] + 1e-16f);
  atomicAdd(agg + (size_t)d * HID + c, alpha * vm[(size_t)s * HID + c]);
}

// exact GELU, output bf16 (feeds the aw-projection WMMA GEMM)
__global__ void k_gelu_bf16(const float* __restrict__ agg, __bf16* __restrict__ out, int n) {
  int i = blockIdx.x * blockDim.x + threadIdx.x;
  if (i < n) {
    float x = agg[i];
    out[i] = (__bf16)(0.5f * x * (1.0f + erff(x * 0.70710678118654752f)));
  }
}

// skip gate + relu + 64->2 classifier
__global__ void k_finalize(const float* __restrict__ obuf, const float* __restrict__ x0,
                           const float* __restrict__ skip, const float* __restrict__ clsw,
                           const float* __restrict__ clsb, float* __restrict__ out, int n) {
  int i = blockIdx.x * blockDim.x + threadIdx.x;
  if (i >= n) return;
  float sg = 1.0f / (1.0f + __expf(-skip[0]));
  float o0 = clsb[0], o1 = clsb[1];
  const float4* op = (const float4*)(obuf + (size_t)i * HID);
  const float4* xp = (const float4*)(x0   + (size_t)i * HID);
#pragma unroll
  for (int j = 0; j < 16; ++j) {
    float4 o = op[j], x = xp[j];
    float y;
    y = fmaxf(sg * o.x + (1.0f - sg) * x.x, 0.0f); o0 += y * clsw[(j*4+0)*2]; o1 += y * clsw[(j*4+0)*2+1];
    y = fmaxf(sg * o.y + (1.0f - sg) * x.y, 0.0f); o0 += y * clsw[(j*4+1)*2]; o1 += y * clsw[(j*4+1)*2+1];
    y = fmaxf(sg * o.z + (1.0f - sg) * x.z, 0.0f); o0 += y * clsw[(j*4+2)*2]; o1 += y * clsw[(j*4+2)*2+1];
    y = fmaxf(sg * o.w + (1.0f - sg) * x.w, 0.0f); o0 += y * clsw[(j*4+3)*2]; o1 += y * clsw[(j*4+3)*2+1];
  }
  out[(size_t)i * 2 + 0] = o0;
  out[(size_t)i * 2 + 1] = o1;
}

// ---------------------------------------------------------------------------
// host-side orchestration
// ---------------------------------------------------------------------------
static inline int gemm_blocks(int M) {
  return ((M >> 4) + 7) / 8;   // 8 waves per block, one 16-row block per wave
}

extern "C" void kernel_launch(void* const* d_in, const int* in_sizes, int n_in,
                              void* d_out, int out_size, void* d_ws, size_t ws_size,
                              hipStream_t stream) {
  (void)n_in; (void)out_size; (void)ws_size;
  const float* flow_x     = (const float*)d_in[0];
  const float* ip_emb     = (const float*)d_in[1];
  const float* port_emb   = (const float*)d_in[2];
  const float* proto_emb  = (const float*)d_in[3];
  const float* flow_lin_w = (const float*)d_in[4];
  const float* flow_lin_b = (const float*)d_in[5];
  const float* kw         = (const float*)d_in[6];
  const float* kb         = (const float*)d_in[7];
  const float* qw         = (const float*)d_in[8];
  const float* qb         = (const float*)d_in[9];
  const float* vw         = (const float*)d_in[10];
  const float* vb         = (const float*)d_in[11];
  const float* aw         = (const float*)d_in[12];
  const float* ab         = (const float*)d_in[13];
  const float* skip       = (const float*)d_in[14];
  const float* a_rel      = (const float*)d_in[15];
  const float* m_rel      = (const float*)d_in[16];
  const float* p_rel      = (const float*)d_in[17];
  const float* cls_w      = (const float*)d_in[18];
  const float* cls_b      = (const float*)d_in[19];
  const int*   edges      = (const int*)  d_in[20];

  const int NF  = in_sizes[0] / FIN;     // 400000 flow nodes
  const int NIp = in_sizes[1] / HID;     // 50000
  const int NPt = in_sizes[2] / HID;     // 65536
  const int NPr = in_sizes[3] / HID;     // 256
  const int NS  = NIp + NPt + NPr;
  const int E   = in_sizes[20] / 12;     // 400000 edges per relation

  // ---- workspace carve-out (256B aligned) --------------------------------
  char* base = (char*)d_ws;
  size_t off = 0;
  auto carve = [&](size_t bytes) { char* p = base + off; off += (bytes + 255) & ~(size_t)255; return p; };
  __bf16* fxbf  = (__bf16*)carve((size_t)NF * FIN * 2);
  float*  x0    = (float*) carve((size_t)NF * HID * 4);
  __bf16* x0bf  = (__bf16*)carve((size_t)NF * HID * 2);
  float*  qf    = (float*) carve((size_t)NF * HID * 4);
  __bf16* embbf = (__bf16*)carve((size_t)NS * HID * 2);
  __bf16* kbf   = (__bf16*)carve((size_t)NS * HID * 2);
  __bf16* vbf   = (__bf16*)carve((size_t)NS * HID * 2);
  float*  krf   = (float*) carve((size_t)NS * HID * 4);
  float*  vmf   = (float*) carve((size_t)NS * HID * 4);
  float*  logit = (float*) carve((size_t)3 * E * 2 * 4);
  unsigned* maxb= (unsigned*)carve((size_t)NF * 2 * 4);
  float*  sumb  = (float*) carve((size_t)NF * 2 * 4);
  float*  agg   = (float*) carve((size_t)NF * HID * 4);
  __bf16* gbf   = (__bf16*)qf;   // alias: qf dead after logit pass
  float*  obuf  = agg;           // alias: agg dead after gelu

  const int T = 256;

  // ---- 1. bf16 conversions ----------------------------------------------
  k_f32_to_bf16<<<(NF * FIN + T - 1) / T, T, 0, stream>>>(flow_x, fxbf, NF * FIN);
  k_f32_to_bf16<<<(NIp * HID + T - 1) / T, T, 0, stream>>>(ip_emb,    embbf,                           NIp * HID);
  k_f32_to_bf16<<<(NPt * HID + T - 1) / T, T, 0, stream>>>(port_emb,  embbf + (size_t)NIp * HID,       NPt * HID);
  k_f32_to_bf16<<<(NPr * HID + T - 1) / T, T, 0, stream>>>(proto_emb, embbf + (size_t)(NIp+NPt)*HID,   NPr * HID);

  // ---- 2. x0 = flow_x @ flow_lin_w + b  (K=32, N=64) ---------------------
  k_gemm_wmma<32, 64, 32, 64, true, true, true><<<gemm_blocks(NF), T, 0, stream>>>(
      fxbf, flow_lin_w, flow_lin_b, x0, x0bf, NF);

  // ---- 3. q_flow = x0 @ qw[0] + qb[0]  (K=64, N=64) ----------------------
  k_gemm_wmma<64, 64, 64, 64, true, true, false><<<gemm_blocks(NF), T, 0, stream>>>(
      x0bf, qw, qb, qf, nullptr, NF);

  // ---- 4. k/v for source types ip(1), port(2), proto(3) ------------------
  const int soff[3] = {0, NIp, NIp + NPt};
  const int scnt[3] = {NIp, NPt, NPr};
  for (int j = 0; j < 3; ++j) {
    int t = j + 1;
    k_gemm_wmma<64, 64, 64, 64, true, false, true><<<gemm_blocks(scnt[j]), T, 0, stream>>>(
        embbf + (size_t)soff[j] * HID, kw + (size_t)t * HID * HID,
        kb + (size_t)t * HID, nullptr, kbf + (size_t)soff[j] * HID, scnt[j]);
    k_gemm_wmma<64, 64, 64, 64, true, false, true><<<gemm_blocks(scnt[j]), T, 0, stream>>>(
        embbf + (size_t)soff[j] * HID, vw + (size_t)t * HID * HID,
        vb + (size_t)t * HID, nullptr, vbf + (size_t)soff[j] * HID, scnt[j]);
  }

  // ---- 5. kr = k @ a_rel[r,h], vm = v @ m_rel[r,h]  (K=N=32, head slices) -
  const int rels[3] = {1, 3, 5};
  for (int j = 0; j < 3; ++j) {
    int r = rels[j];
    for (int h = 0; h < HEADS; ++h) {
      const float* Ar = a_rel + ((size_t)r * HEADS + h) * DH * DH;
      const float* Mr = m_rel + ((size_t)r * HEADS + h) * DH * DH;
      k_gemm_wmma<32, 32, 64, 64, false, true, false><<<gemm_blocks(scnt[j]), T, 0, stream>>>(
          kbf + (size_t)soff[j] * HID + h * DH, Ar, nullptr,
          krf + (size_t)soff[j] * HID + h * DH, nullptr, scnt[j]);
      k_gemm_wmma<32, 32, 64, 64, false, true, false><<<gemm_blocks(scnt[j]), T, 0, stream>>>(
          vbf + (size_t)soff[j] * HID + h * DH, Mr, nullptr,
          vmf + (size_t)soff[j] * HID + h * DH, nullptr, scnt[j]);
    }
  }

  // ---- 6. segment softmax + aggregation into flow nodes ------------------
  hipMemsetAsync(maxb, 0, (size_t)NF * 2 * 4, stream);   // 0 == ord_enc lower bound
  hipMemsetAsync(sumb, 0, (size_t)NF * 2 * 4, stream);
  hipMemsetAsync(agg,  0, (size_t)NF * HID * 4, stream);

  for (int j = 0; j < 3; ++j) {
    int r = rels[j];
    const int* src = edges + (size_t)r * 2 * E;
    const int* dst = src + E;
    k_edge_logit<<<(E * 2 + T - 1) / T, T, 0, stream>>>(
        src, dst, qf, krf + (size_t)soff[j] * HID, p_rel + (size_t)r * HEADS,
        logit + (size_t)j * E * 2, maxb, E);
  }
  for (int j = 0; j < 3; ++j) {
    const int* dst = edges + (size_t)rels[j] * 2 * E + E;
    k_edge_exp<<<(E * 2 + T - 1) / T, T, 0, stream>>>(
        dst, logit + (size_t)j * E * 2, maxb, sumb, E);
  }
  for (int j = 0; j < 3; ++j) {
    const int* src = edges + (size_t)rels[j] * 2 * E;
    const int* dst = src + E;
    k_edge_agg<<<(int)(((long long)E * HID + T - 1) / T), T, 0, stream>>>(
        src, dst, logit + (size_t)j * E * 2, sumb,
        vmf + (size_t)soff[j] * HID, agg, E);
  }

  // ---- 7. o = gelu(agg) @ aw[0] + ab[0]; skip-gate; classifier -----------
  k_gelu_bf16<<<(NF * HID + T - 1) / T, T, 0, stream>>>(agg, gbf, NF * HID);
  k_gemm_wmma<64, 64, 64, 64, true, true, false><<<gemm_blocks(NF), T, 0, stream>>>(
      gbf, aw, ab, obuf, nullptr, NF);
  k_finalize<<<(NF + T - 1) / T, T, 0, stream>>>(
      obuf, x0, skip, cls_w, cls_b, (float*)d_out, NF);
}
